// joint_test_90598040142522
// MI455X (gfx1250) — matrix operand order repaired and checked by reading, not compiled, not run on hardware
//
#include <hip/hip_runtime.h>

typedef __attribute__((ext_vector_type(16))) __bf16 v16bf;
typedef __attribute__((ext_vector_type(8)))  float  v8f;
typedef __attribute__((ext_vector_type(4)))  unsigned short us4;

#define NROWS 65536
#define KDIM  512
#define A3    64
#define CWIN  16
#define DTOW  8
#define PLEN  96
#define OUTW  (DTOW * PLEN)   // 768

// native f32 -> bf16 (RTE) via hardware convert
__device__ __forceinline__ unsigned short f2bfbits(float f) {
  return __builtin_bit_cast(unsigned short, (__bf16)f);
}

union BF16Frag {
  v16bf v;
  unsigned short u[16];
};

// 16 contiguous halves (32B, 32B-aligned) -> one WMMA operand (2x b128 loads)
__device__ __forceinline__ v16bf load_frag(const unsigned short* __restrict__ p) {
  BF16Frag f;
#pragma unroll
  for (int e = 0; e < 16; ++e) f.u[e] = p[e];
  return f.v;
}

// ---------------------------------------------------------------------------
// Weight pre-pack kernels: reorder into WMMA B-fragment order
// layout: [...][kb][nt][lane(32)][e(16)] bf16 ; lane = (col%16) + 16*(k/16 % 2)
// ---------------------------------------------------------------------------
__global__ __launch_bounds__(256) void pack_S(const float* __restrict__ S,
                                              unsigned short* __restrict__ p) {
  int j = blockIdx.x * 256 + threadIdx.x;          // < 16*4*32*16 = 32768
  int e = j & 15, lane = (j >> 4) & 31, nt = (j >> 9) & 3, kb = j >> 11;
  int k = kb * 32 + (lane >> 4) * 16 + e;
  int col = nt * 16 + (lane & 15);
  p[j] = f2bfbits(S[(size_t)k * A3 + col]);
}

__global__ __launch_bounds__(256) void pack_W1(const float* __restrict__ W1,
                                               unsigned short* __restrict__ p) {
  int j = blockIdx.x * 256 + threadIdx.x;          // < 16*2*4*32*16 = 65536
  int e = j & 15, lane = (j >> 4) & 31, nt = (j >> 9) & 3;
  int kb = (j >> 11) & 1, d = j >> 12;
  int k = kb * 32 + (lane >> 4) * 16 + e;
  int col = nt * 16 + (lane & 15);
  p[j] = f2bfbits(W1[(size_t)d * 64 * 64 + (size_t)k * 64 + col]);
}

__global__ __launch_bounds__(256) void pack_W2(const float* __restrict__ W2,
                                               unsigned short* __restrict__ p) {
  int j = blockIdx.x * 256 + threadIdx.x;          // < 16*2*6*32*16 = 98304
  int e = j & 15, lane = (j >> 4) & 31;
  int idx512 = j >> 9;                             // 0..191
  int nt = idx512 % 6, g = idx512 / 6;             // g: 0..31
  int kb = g & 1, d = g >> 1;
  int k = kb * 32 + (lane >> 4) * 16 + e;
  int col = nt * 16 + (lane & 15);
  p[j] = f2bfbits(W2[(size_t)d * 64 * 96 + (size_t)k * 96 + col]);
}

// ---------------------------------------------------------------------------
// K1: t[N,64] = x[N,512] @ S[512,64]  (bf16 WMMA, f32 accum)
// block 256 thr = 8 waves; tile 128x64; B fragments streamed from packed L2
// ---------------------------------------------------------------------------
__global__ __launch_bounds__(256) void k1_gemm_t(const float* __restrict__ x,
                                                 const unsigned short* __restrict__ packS,
                                                 float* __restrict__ t) {
  __shared__ unsigned short lx[128 * 32];
  const int tid  = threadIdx.x;
  const int wave = tid >> 5;
  const int lane = tid & 31;
  const int r    = lane & 15;
  const int kh   = lane >> 4;
  const int mblk = blockIdx.x * 128;

  v8f acc[4] = {};

  for (int k0 = 0; k0 < KDIM; k0 += 32) {
    // stage A tile as bf16: 128x32, float4 reads -> ushort4 LDS writes
    for (int i = tid; i < 1024; i += 256) {
      int row = i >> 3, colq = i & 7;
      float4 v = *(const float4*)&x[(size_t)(mblk + row) * KDIM + k0 + colq * 4];
      us4 b; b.x = f2bfbits(v.x); b.y = f2bfbits(v.y);
      b.z = f2bfbits(v.z); b.w = f2bfbits(v.w);
      *(us4*)&lx[row * 32 + colq * 4] = b;
    }
    __syncthreads();

    BF16Frag a;
    const unsigned short* arow = &lx[(16 * wave + r) * 32];
#pragma unroll
    for (int e = 0; e < 16; ++e) {
      int kk = e + 16 * kh;
      a.u[e] = arow[kk];
    }
    const int kb = k0 >> 5;
#pragma unroll
    for (int nt = 0; nt < 4; ++nt) {
      v16bf b = load_frag(packS + ((size_t)(kb * 4 + nt) * 32 + lane) * 16);
      acc[nt] = __builtin_amdgcn_wmma_f32_16x16x32_bf16(
          false, a.v, false, b, (short)0, acc[nt], false, false);
    }
    __syncthreads();
  }
#pragma unroll
  for (int nt = 0; nt < 4; ++nt)
#pragma unroll
    for (int v = 0; v < 8; ++v)
      t[(size_t)(mblk + 16 * wave + v + 8 * kh) * A3 + 16 * nt + r] = acc[nt][v];
}

// ---------------------------------------------------------------------------
// K2a: row-wise L2 normalize t in place. one wave per 64-wide row.
// ---------------------------------------------------------------------------
__global__ __launch_bounds__(256) void k2_normalize(float* __restrict__ t) {
  const int wave = threadIdx.x >> 5;
  const int lane = threadIdx.x & 31;
  const int row  = blockIdx.x * 8 + wave;
  float2 v = *(const float2*)&t[(size_t)row * A3 + lane * 2];
  float ss = v.x * v.x + v.y * v.y;
#pragma unroll
  for (int off = 16; off > 0; off >>= 1)
    ss += __shfl_xor(ss, off, 32);
  float inv = 1.0f / fmaxf(sqrtf(ss), 1e-12f);
  v.x *= inv; v.y *= inv;
  *(float2*)&t[(size_t)row * A3 + lane * 2] = v;
}

// ---------------------------------------------------------------------------
// K2b: rep[i,a] = sum_j lw[j,a]*t[idx(i,j),a] + sum_j lb[j,a]
// halo tile staged with ASYNCcnt-tracked global->LDS DMA (gfx1250 path).
// Negative halo rows clamp to row 0: provably never read (block 0 uses
// idx=min(j,i)>=0 -> local index >= 15).
// ---------------------------------------------------------------------------
__global__ __launch_bounds__(256) void k2b_windowsum(const float* __restrict__ t,
                                                     const float* __restrict__ lw,
                                                     const float* __restrict__ lb,
                                                     float* __restrict__ rep) {
  __shared__ float tile[(64 + CWIN - 1) * A3];  // 79 x 64
  __shared__ float slw[CWIN * A3];
  __shared__ float sbsum[A3];
  const int tid = threadIdx.x;
  const int i0  = blockIdx.x * 64;

  // async copy 79x64 f32 tile, 16B granules
  for (int i = tid; i < (64 + CWIN - 1) * (A3 / 4); i += 256) {  // 1264 quads
    int rl = i >> 4, cq = i & 15;
    int rg = i0 - (CWIN - 1) + rl;
    if (rg < 0) rg = 0;
    unsigned long long ga =
        (unsigned long long)(uintptr_t)&t[(size_t)rg * A3 + cq * 4];
    unsigned lds_off = (unsigned)(uintptr_t)&tile[rl * A3 + cq * 4];
    asm volatile("global_load_async_to_lds_b128 %0, %1, off"
                 :: "v"(lds_off), "v"(ga) : "memory");
  }
  for (int i = tid; i < CWIN * A3; i += 256) slw[i] = lw[i];
  if (tid < A3) {
    float s = 0.f;
#pragma unroll
    for (int j = 0; j < CWIN; ++j) s += lb[j * A3 + tid];
    sbsum[tid] = s;
  }
  asm volatile("s_wait_asynccnt 0x0" ::: "memory");
  __syncthreads();

  const int col = tid & 63;
  for (int s = 0; s < 16; ++s) {
    int rl = (tid >> 6) + 4 * s;  // 0..63
    int i  = i0 + rl;
    float acc = sbsum[col];
#pragma unroll
    for (int j = 0; j < CWIN; ++j) {
      int idx = (i >= CWIN - 1) ? (i - (CWIN - 1) + j) : (j < i ? j : i);
      acc += slw[j * A3 + col] * tile[(idx - (i0 - (CWIN - 1))) * A3 + col];
    }
    rep[(size_t)i * A3 + col] = acc;
  }
}

// ---------------------------------------------------------------------------
// K3: h[d] = rep @ W1[d] + b1[d]   (bf16 WMMA). grid = (N/128, D)
// ---------------------------------------------------------------------------
__global__ __launch_bounds__(256) void k3_tower1(const float* __restrict__ rep,
                                                 const unsigned short* __restrict__ packW1,
                                                 const float* __restrict__ b1,
                                                 float* __restrict__ h) {
  __shared__ unsigned short lrep[128 * 64];
  const int tid  = threadIdx.x;
  const int wave = tid >> 5;
  const int lane = tid & 31;
  const int r    = lane & 15;
  const int kh   = lane >> 4;
  const int mblk = blockIdx.x * 128;
  const int d    = blockIdx.y;

  for (int i = tid; i < 2048; i += 256) {
    int row = i >> 4, colq = i & 15;
    float4 v = *(const float4*)&rep[(size_t)(mblk + row) * A3 + colq * 4];
    us4 b; b.x = f2bfbits(v.x); b.y = f2bfbits(v.y);
    b.z = f2bfbits(v.z); b.w = f2bfbits(v.w);
    *(us4*)&lrep[row * 64 + colq * 4] = b;
  }
  __syncthreads();

  v8f acc[4] = {};
#pragma unroll
  for (int kb = 0; kb < 2; ++kb) {
    BF16Frag a;
    const unsigned short* arow = &lrep[(16 * wave + r) * 64 + kb * 32];
#pragma unroll
    for (int e = 0; e < 16; ++e) {
      int kk = e + 16 * kh;
      a.u[e] = arow[kk];
    }
#pragma unroll
    for (int nt = 0; nt < 4; ++nt) {
      v16bf b = load_frag(packW1 + ((size_t)((d * 2 + kb) * 4 + nt) * 32 + lane) * 16);
      acc[nt] = __builtin_amdgcn_wmma_f32_16x16x32_bf16(
          false, a.v, false, b, (short)0, acc[nt], false, false);
    }
  }
#pragma unroll
  for (int nt = 0; nt < 4; ++nt) {
    float bias = b1[d * A3 + 16 * nt + r];
#pragma unroll
    for (int v = 0; v < 8; ++v)
      h[((size_t)d * NROWS + mblk + 16 * wave + v + 8 * kh) * A3 + 16 * nt + r] =
          acc[nt][v] + bias;
  }
}

// ---------------------------------------------------------------------------
// K4a: coalesced partial BN stats. grid (64, D); block reduces 1024 rows.
// ---------------------------------------------------------------------------
__global__ __launch_bounds__(256) void k4a_partial(const float* __restrict__ h,
                                                   float* __restrict__ psum,
                                                   float* __restrict__ psq) {
  __shared__ float4 ssum[256], ssq[256];
  const int tid = threadIdx.x;
  const int rl  = tid >> 4;      // 0..15
  const int cq  = tid & 15;      // col quad 0..15
  const int blk = blockIdx.x;    // 0..63
  const int d   = blockIdx.y;
  const float* base = h + ((size_t)d * NROWS + (size_t)blk * 1024) * A3;
  float4 s = {0.f, 0.f, 0.f, 0.f}, q = {0.f, 0.f, 0.f, 0.f};
  for (int rr = rl; rr < 1024; rr += 16) {
    float4 v = *(const float4*)(base + (size_t)rr * A3 + cq * 4);
    s.x += v.x; s.y += v.y; s.z += v.z; s.w += v.w;
    q.x += v.x * v.x; q.y += v.y * v.y; q.z += v.z * v.z; q.w += v.w * v.w;
  }
  ssum[tid] = s; ssq[tid] = q;
  __syncthreads();
  for (int off = 8; off > 0; off >>= 1) {
    if (rl < off) {
      float4 a = ssum[tid], b = ssum[tid + off * 16];
      a.x += b.x; a.y += b.y; a.z += b.z; a.w += b.w; ssum[tid] = a;
      float4 c = ssq[tid], e = ssq[tid + off * 16];
      c.x += e.x; c.y += e.y; c.z += e.z; c.w += e.w; ssq[tid] = c;
    }
    __syncthreads();
  }
  if (rl == 0) {
    *(float4*)&psum[(size_t)(d * 64 + blk) * A3 + cq * 4] = ssum[tid];
    *(float4*)&psq [(size_t)(d * 64 + blk) * A3 + cq * 4] = ssq[tid];
  }
}

// K4b: fold 64 partials/channel -> BN scale/shift (deterministic, tiny)
__global__ __launch_bounds__(256) void k4b_finalize(const float* __restrict__ psum,
                                                    const float* __restrict__ psq,
                                                    const float* __restrict__ gamma,
                                                    const float* __restrict__ beta,
                                                    float* __restrict__ scale,
                                                    float* __restrict__ shift) {
  for (int ch = threadIdx.x; ch < DTOW * A3; ch += 256) {
    int d = ch >> 6, c = ch & 63;
    float s = 0.f, q = 0.f;
    for (int b = 0; b < 64; ++b) {
      s += psum[(size_t)(d * 64 + b) * A3 + c];
      q += psq [(size_t)(d * 64 + b) * A3 + c];
    }
    float mu  = s * (1.0f / NROWS);
    float var = fmaxf(q * (1.0f / NROWS) - mu * mu, 0.f);
    float sc  = gamma[ch] * rsqrtf(var + 1e-5f);
    scale[ch] = sc;
    shift[ch] = beta[ch] - mu * sc;
  }
}

// ---------------------------------------------------------------------------
// K5: y = ELU(h*scale+shift) @ W2[d] + b2[d], scattered to out[n, p*8+d]
// grid = (N/128, D)
// ---------------------------------------------------------------------------
__global__ __launch_bounds__(256) void k5_tower2(const float* __restrict__ h,
                                                 const float* __restrict__ scale,
                                                 const float* __restrict__ shift,
                                                 const unsigned short* __restrict__ packW2,
                                                 const float* __restrict__ b2,
                                                 float* __restrict__ out) {
  __shared__ unsigned short lh[128 * 64];
  const int tid  = threadIdx.x;
  const int wave = tid >> 5;
  const int lane = tid & 31;
  const int r    = lane & 15;
  const int kh   = lane >> 4;
  const int mblk = blockIdx.x * 128;
  const int d    = blockIdx.y;

  for (int i = tid; i < 2048; i += 256) {
    int row = i >> 4, colq = i & 15;
    float4 v  = *(const float4*)&h[((size_t)d * NROWS + mblk + row) * A3 + colq * 4];
    float4 sc = *(const float4*)&scale[d * A3 + colq * 4];
    float4 sh = *(const float4*)&shift[d * A3 + colq * 4];
    float a0 = v.x * sc.x + sh.x, a1 = v.y * sc.y + sh.y;
    float a2 = v.z * sc.z + sh.z, a3 = v.w * sc.w + sh.w;
    a0 = (a0 > 0.f) ? a0 : expm1f(a0);
    a1 = (a1 > 0.f) ? a1 : expm1f(a1);
    a2 = (a2 > 0.f) ? a2 : expm1f(a2);
    a3 = (a3 > 0.f) ? a3 : expm1f(a3);
    us4 b; b.x = f2bfbits(a0); b.y = f2bfbits(a1);
    b.z = f2bfbits(a2); b.w = f2bfbits(a3);
    *(us4*)&lh[row * 64 + colq * 4] = b;
  }
  __syncthreads();

  v8f acc[6] = {};
#pragma unroll
  for (int kb = 0; kb < 2; ++kb) {
    BF16Frag a;
    const unsigned short* arow = &lh[(16 * wave + r) * 64 + kb * 32];
#pragma unroll
    for (int e = 0; e < 16; ++e) {
      int kk = e + 16 * kh;
      a.u[e] = arow[kk];
    }
#pragma unroll
    for (int nt = 0; nt < 6; ++nt) {
      v16bf b = load_frag(packW2 + ((size_t)((d * 2 + kb) * 6 + nt) * 32 + lane) * 16);
      acc[nt] = __builtin_amdgcn_wmma_f32_16x16x32_bf16(
          false, a.v, false, b, (short)0, acc[nt], false, false);
    }
  }
#pragma unroll
  for (int nt = 0; nt < 6; ++nt) {
    int p = 16 * nt + r;
    float bias = b2[d * PLEN + p];
#pragma unroll
    for (int v = 0; v < 8; ++v) {
      int row = mblk + 16 * wave + v + 8 * kh;
      out[(size_t)row * OUTW + p * DTOW + d] = acc[nt][v] + bias;
    }
  }
}

// ---------------------------------------------------------------------------
extern "C" void kernel_launch(void* const* d_in, const int* in_sizes, int n_in,
                              void* d_out, int out_size, void* d_ws, size_t ws_size,
                              hipStream_t stream) {
  const float* x     = (const float*)d_in[0];
  const float* S     = (const float*)d_in[1];
  const float* lw    = (const float*)d_in[2];
  const float* lb    = (const float*)d_in[3];
  const float* W1    = (const float*)d_in[4];
  const float* b1    = (const float*)d_in[5];
  const float* gamma = (const float*)d_in[6];
  const float* beta  = (const float*)d_in[7];
  const float* W2    = (const float*)d_in[8];
  const float* b2    = (const float*)d_in[9];
  // d_in[10] = flag (reference path flag=1), d_in[11] = label (unused)

  float* out     = (float*)d_out;
  float* rep_out = out + (size_t)NROWS * OUTW;   // second tuple output: rep [N,64]

  float* t     = (float*)d_ws;                   // [N,64]        16 MB
  float* h     = t + (size_t)NROWS * A3;         // [D,N,64]     134 MB
  float* scale = h + (size_t)DTOW * NROWS * A3;  // [512]
  float* shift = scale + DTOW * A3;              // [512]
  float* psum  = shift + DTOW * A3;              // [512*64]
  float* psq   = psum + DTOW * A3 * 64;          // [512*64]
  unsigned short* packS  = (unsigned short*)(psq + DTOW * A3 * 64);  // 32768 bf16
  unsigned short* packW1 = packS + 32768;                            // 65536 bf16
  unsigned short* packW2 = packW1 + 65536;                           // 98304 bf16

  pack_S  <<<128, 256, 0, stream>>>(S, packS);
  pack_W1 <<<256, 256, 0, stream>>>(W1, packW1);
  pack_W2 <<<384, 256, 0, stream>>>(W2, packW2);

  k1_gemm_t    <<<NROWS / 128, 256, 0, stream>>>(x, packS, t);
  k2_normalize <<<NROWS / 8,   256, 0, stream>>>(t);
  k2b_windowsum<<<NROWS / 64,  256, 0, stream>>>(t, lw, lb, rep_out);
  k3_tower1    <<<dim3(NROWS / 128, DTOW), 256, 0, stream>>>(rep_out, packW1, b1, h);
  k4a_partial  <<<dim3(64, DTOW), 256, 0, stream>>>(h, psum, psq);
  k4b_finalize <<<1, 256, 0, stream>>>(psum, psq, gamma, beta, scale, shift);
  k5_tower2    <<<dim3(NROWS / 128, DTOW), 256, 0, stream>>>(h, scale, shift, packW2, b2, out);
}
